// MultiHeadLatentAttentionSimple_37349035606741
// MI455X (gfx1250) — compile-verified, matrix operand order checked by reading, not run on hardware
//
#include <hip/hip_runtime.h>

// ---------------------------------------------------------------------------
// MLA forward for MI455X (gfx1250, wave32, WMMA + async/TDM data movement).
//   Q = q@Wq+bq ; lat = k@Wc+bc ; K = lat@Wk+bk ; V = lat@Wv+bv
//   attn = softmax(QK^T/sqrt(128)) @ V ; out = attn@Wo + bo
// Matmuls: v_wmma_f32_16x16x32_bf16.
// A-tile staging: global_load_async_to_lds_b128 (ASYNCcnt).
// Attention K-tile: tensor_load_to_lds with LDS padding (TENSORcnt).
// ---------------------------------------------------------------------------

typedef __attribute__((ext_vector_type(8)))  __bf16 bf16x8;
typedef __attribute__((ext_vector_type(16))) __bf16 bf16x16;
typedef __attribute__((ext_vector_type(8)))  float  v8f;
typedef unsigned int u32x4 __attribute__((ext_vector_type(4)));
typedef int          i32x8 __attribute__((ext_vector_type(8)));
typedef int          i32x4 __attribute__((ext_vector_type(4)));

#define SEQ    2048
#define DMODEL 2048
#define NHEAD  16
#define DKV    128
#define LATENT 512
#define BATCH  4
#define MROWS  (BATCH * SEQ)   // 8192

#define MLA_USE_TDM 1

// 16-bit A-matrix fragment (16x32) per ISA table:
//   lanes 0-15 : row M = lane,    K = {0..7, 16..23}
//   lanes 16-31: row M = lane-16, K = {8..15, 24..31}
// -> two contiguous 16-byte reads at k-offsets (half*8) and (16+half*8).
__device__ __forceinline__ bf16x16 load_frag(const __bf16* rowptr, int half) {
  bf16x8 lo = *(const bf16x8*)(rowptr + half * 8);
  bf16x8 hi = *(const bf16x8*)(rowptr + 16 + half * 8);
  return __builtin_shufflevector(lo, hi, 0, 1, 2, 3, 4, 5, 6, 7,
                                         8, 9, 10, 11, 12, 13, 14, 15);
}

__device__ __forceinline__ v8f wmma_bf16(bf16x16 a, bf16x16 b, v8f c) {
  return __builtin_amdgcn_wmma_f32_16x16x32_bf16(false, a, false, b,
                                                 (short)0, c, false, false);
}

// Async global->LDS copy of 16 bytes per lane (VGLOBAL async, ASYNCcnt).
__device__ __forceinline__ void async_copy_b128(unsigned int lds_byte_off,
                                                const void* gsrc) {
  asm volatile("global_load_async_to_lds_b128 %0, %1, off"
               :: "v"(lds_byte_off), "v"(gsrc) : "memory");
}
__device__ __forceinline__ void wait_asynccnt0() {
  asm volatile("s_wait_asynccnt 0x0" ::: "memory");
}
__device__ __forceinline__ void wait_tensorcnt0() {
#if __has_builtin(__builtin_amdgcn_s_wait_tensorcnt)
  __builtin_amdgcn_s_wait_tensorcnt(0);
#else
  asm volatile("s_wait_tensorcnt 0x0" ::: "memory");
#endif
}

// ---------------------------------------------------------------------------
// fp32 -> bf16 conversion
// ---------------------------------------------------------------------------
__global__ void cvt_f32_bf16(const float* __restrict__ src,
                             __bf16* __restrict__ dst, size_t n) {
  size_t i = (size_t)blockIdx.x * blockDim.x + threadIdx.x;
  size_t stride = (size_t)gridDim.x * blockDim.x;
  for (; i < n; i += stride) dst[i] = (__bf16)src[i];
}

// ---------------------------------------------------------------------------
// Tiled WMMA GEMM: C[M,N] = A[M,K] @ W[K,N] + bias[N]
// block = 256 threads (8 waves), block tile 128x128, K-step 32.
// wave (wm,wn): wm = (wave>>1)*32, wn = (wave&1)*64 -> 2x4 tiles of 16x16.
// A tile staged by async global->LDS; W tile staged transposed manually.
// ---------------------------------------------------------------------------
__global__ __launch_bounds__(256) void gemm_bf16_wmma(
    const __bf16* __restrict__ A, const __bf16* __restrict__ W,
    const float* __restrict__ bias, float* __restrict__ Cf,
    __bf16* __restrict__ Cb, int M, int N, int K) {
  __shared__ __bf16 As[128][40];    // [m][k], 80B row stride (16B aligned)
  __shared__ __bf16 BsT[128][40];   // [n][k]

  const int tid  = threadIdx.x;
  const int wave = tid >> 5;
  const int lane = tid & 31;
  const int half = lane >> 4;
  const int l16  = lane & 15;
  const int bm = blockIdx.y * 128;
  const int bn = blockIdx.x * 128;
  const int wm = (wave >> 1) * 32;
  const int wn = (wave & 1) * 64;

  v8f acc[2][4];
#pragma unroll
  for (int i = 0; i < 2; ++i)
#pragma unroll
    for (int j = 0; j < 4; ++j) acc[i][j] = (v8f)(0.0f);

  const int ar = tid >> 1;                 // A tile row 0..127
  const int ac = (tid & 1) * 16;           // A tile col 0 or 16
  const int wk = tid >> 3;                 // W tile k-row 0..31
  const int wc = (tid & 7) * 16;           // W tile col 0..112
  const unsigned int lds_a = (unsigned int)(uintptr_t)&As[ar][ac];

  for (int k0 = 0; k0 < K; k0 += 32) {
    __syncthreads();   // WAR: previous iteration's fragment reads done
    {
      // A tile: direct async global->LDS (no VGPR round trip)
      const __bf16* srcA = A + (size_t)(bm + ar) * K + k0 + ac;
      async_copy_b128(lds_a, srcA);
      async_copy_b128(lds_a + 16, srcA + 8);
      if (k0 + 32 < K) __builtin_prefetch(srcA + 32, 0, 0);

      // W tile: load rows, scatter transposed into BsT[n][k]
      const __bf16* srcW = W + (size_t)(k0 + wk) * N + bn + wc;
      bf16x8 w0 = *(const bf16x8*)srcW;
      bf16x8 w1 = *(const bf16x8*)(srcW + 8);
#pragma unroll
      for (int i = 0; i < 8; ++i) {
        BsT[wc + i][wk]     = w0[i];
        BsT[wc + 8 + i][wk] = w1[i];
      }
    }
    wait_asynccnt0();
    __syncthreads();

    bf16x16 af[2], bfg[4];
#pragma unroll
    for (int i = 0; i < 2; ++i) af[i] = load_frag(&As[wm + i * 16 + l16][0], half);
#pragma unroll
    for (int j = 0; j < 4; ++j) bfg[j] = load_frag(&BsT[wn + j * 16 + l16][0], half);
#pragma unroll
    for (int i = 0; i < 2; ++i)
#pragma unroll
      for (int j = 0; j < 4; ++j) acc[i][j] = wmma_bf16(af[i], bfg[j], acc[i][j]);
  }

  // epilogue: C layout VGPR r -> row half*8+r, col = l16 within tile
#pragma unroll
  for (int j = 0; j < 4; ++j) {
    const int col = bn + wn + j * 16 + l16;
    const float bv = bias[col];
#pragma unroll
    for (int i = 0; i < 2; ++i) {
      const int row0 = bm + wm + i * 16 + half * 8;
      if (Cf) {
#pragma unroll
        for (int r = 0; r < 8; ++r)
          Cf[(size_t)(row0 + r) * N + col] = acc[i][j][r] + bv;
      } else {
#pragma unroll
        for (int r = 0; r < 8; ++r)
          Cb[(size_t)(row0 + r) * N + col] = (__bf16)(acc[i][j][r] + bv);
      }
    }
  }
}

// ---------------------------------------------------------------------------
// Flash attention, one (b,h,128-query-row) tile per block.
// 8 waves x 16 query rows; 32 keys per iteration; all matmuls via WMMA.
// K-tile fetched by the Tensor Data Mover with LDS padding to stride 144.
// ---------------------------------------------------------------------------
__global__ __launch_bounds__(256) void mla_attention(
    const __bf16* __restrict__ Q, const __bf16* __restrict__ Kp,
    const __bf16* __restrict__ Vp, __bf16* __restrict__ O) {
  __shared__ __bf16 Ks[32][144];       // [key][d]   (B-layout for QK^T)
  __shared__ __bf16 Vs[128][40];       // [d][key]   (B-layout for P·V)
  __shared__ float  Ss[8][16][33];     // per-wave score tile
  __shared__ __bf16 Ps[8][16][40];     // per-wave probability tile
  __shared__ float  Rs[8][16];         // per-row rescale / final l broadcast

  const int tid  = threadIdx.x;
  const int wave = tid >> 5;
  const int lane = tid & 31;
  const int half = lane >> 4;
  const int l16  = lane & 15;
  const int h = blockIdx.y;
  const int b = blockIdx.z;
  const float scale = 0.08838834764831845f;  // 1/sqrt(128)

  // preload this wave's Q fragments (16 rows x 128 d = 4 K-windows)
  bf16x16 qf[4];
  {
    const int qrow = blockIdx.x * 128 + wave * 16 + l16;
    const __bf16* qptr = Q + ((size_t)(b * SEQ + qrow)) * DMODEL + h * DKV;
#pragma unroll
    for (int kw = 0; kw < 4; ++kw) qf[kw] = load_frag(qptr + kw * 32, half);
  }

  v8f acc[8];
#pragma unroll
  for (int nt = 0; nt < 8; ++nt) acc[nt] = (v8f)(0.0f);
  float m_i = -1e30f, l_i = 0.0f;

  const int kvr = tid >> 3;        // key row in tile 0..31
  const int kvc = (tid & 7) * 16;  // d col 0..112

  for (int j = 0; j < SEQ / 32; ++j) {
    __syncthreads();  // WAR on Ks/Vs/Ps
#if MLA_USE_TDM && __has_builtin(__builtin_amdgcn_tensor_load_to_lds)
    if (wave == 0) {
      // D# group0: count=1 | lds_addr | global_addr(57b) | type=2
      unsigned long long ga =
          (unsigned long long)(uintptr_t)(Kp +
              ((size_t)(b * SEQ + j * 32)) * DMODEL + h * DKV);
      u32x4 g0;
      g0.x = 1u;                                   // count=1 user descriptor
      g0.y = (unsigned int)(uintptr_t)&Ks[0][0];   // lds_addr (byte offset)
      g0.z = (unsigned int)ga;                     // global_addr[31:0]
      g0.w = (unsigned int)((ga >> 32) & 0x1FFFFFFu) | (2u << 30);  // type=2
      // D# group1: data_size=2B, pad every 64 DW by 8 DW (row 256B -> 288B)
      i32x8 g1;
      g1[0] = (int)((1u << 16)            // data_size code 1 (2 bytes)
                  | (1u << 20)            // pad_enable
                  | (5u << 22)            // pad_interval: 64 DWORDs
                  | (7u << 25));          // pad_amount:   8 DWORDs
      g1[1] = (int)(128u << 16);          // tensor_dim0 = 128 (lo16)
      g1[2] = (int)(8192u << 16);         // dim0 hi=0 | tensor_dim1 = 8192 lo16
      g1[3] = (int)(128u << 16);          // dim1 hi=0 | tile_dim0 = 128
      g1[4] = (int)32u;                   // tile_dim1 = 32 | tile_dim2 = 0
      g1[5] = (int)2048u;                 // tensor_dim0_stride = 2048 (lo32)
      g1[6] = 0;                          // stride hi | dim1_stride lo
      g1[7] = 0;
      i32x4 gz4 = {0, 0, 0, 0};
      i32x8 gz8 = {0, 0, 0, 0, 0, 0, 0, 0};
      // 6-arg toolchain variant: (g0, g1, g2, g3, g4, cpol)
      __builtin_amdgcn_tensor_load_to_lds(g0, g1, gz4, gz4, gz8, 0);
    }
#else
    {
      size_t kbase = ((size_t)(b * SEQ + j * 32 + kvr)) * DMODEL + h * DKV + kvc;
      bf16x8 k0 = *(const bf16x8*)(Kp + kbase);
      bf16x8 k1 = *(const bf16x8*)(Kp + kbase + 8);
      *(bf16x8*)&Ks[kvr][kvc]     = k0;
      *(bf16x8*)&Ks[kvr][kvc + 8] = k1;
    }
#endif
    {
      // V tile transposed into Vs[d][key] (manual: TDM cannot transpose)
      size_t vbase = ((size_t)(b * SEQ + j * 32 + kvr)) * DMODEL + h * DKV + kvc;
      bf16x8 v0 = *(const bf16x8*)(Vp + vbase);
      bf16x8 v1 = *(const bf16x8*)(Vp + vbase + 8);
#pragma unroll
      for (int i = 0; i < 8; ++i) {
        Vs[kvc + i][kvr]     = v0[i];
        Vs[kvc + 8 + i][kvr] = v1[i];
      }
    }
#if MLA_USE_TDM && __has_builtin(__builtin_amdgcn_tensor_load_to_lds)
    if (wave == 0) wait_tensorcnt0();
#endif
    __syncthreads();

    // S = Q K^T * scale : 16x32 per wave (2 N-tiles, 4 K-windows)
#pragma unroll
    for (int jt = 0; jt < 2; ++jt) {
      v8f sc = (v8f)(0.0f);
#pragma unroll
      for (int kw = 0; kw < 4; ++kw) {
        bf16x16 kf = load_frag(&Ks[jt * 16 + l16][kw * 32], half);
        sc = wmma_bf16(qf[kw], kf, sc);
      }
#pragma unroll
      for (int r = 0; r < 8; ++r)
        Ss[wave][half * 8 + r][jt * 16 + l16] = sc[r] * scale;
    }
    __syncthreads();

    // online softmax: lane owns row l16 (both halves duplicate the math)
    {
      float srow[32];
      float mloc = m_i;
#pragma unroll
      for (int c = 0; c < 32; ++c) {
        srow[c] = Ss[wave][l16][c];
        mloc = fmaxf(mloc, srow[c]);
      }
      const float resc = __expf(m_i - mloc);
      float psum = 0.0f;
#pragma unroll
      for (int c = 0; c < 32; ++c) {
        float p = __expf(srow[c] - mloc);
        psum += p;
        if (half == 0) Ps[wave][l16][c] = (__bf16)p;
      }
      l_i = l_i * resc + psum;
      m_i = mloc;
      if (half == 0) Rs[wave][l16] = resc;
    }
    __syncthreads();

    // rescale accumulators and add P·V (8 N-tiles over d=128)
    {
      float rfac[8];
#pragma unroll
      for (int r = 0; r < 8; ++r) rfac[r] = Rs[wave][half * 8 + r];
      bf16x16 pf = load_frag(&Ps[wave][l16][0], half);
#pragma unroll
      for (int nt = 0; nt < 8; ++nt) {
#pragma unroll
        for (int r = 0; r < 8; ++r) acc[nt][r] *= rfac[r];
        bf16x16 vf = load_frag(&Vs[nt * 16 + l16][0], half);
        acc[nt] = wmma_bf16(pf, vf, acc[nt]);
      }
    }
  }

  // broadcast final row sums, normalize, store
  __syncthreads();
  if (half == 0) Rs[wave][l16] = l_i;
  __syncthreads();
  {
    const int orow0 = blockIdx.x * 128 + wave * 16 + half * 8;
#pragma unroll
    for (int r = 0; r < 8; ++r) {
      const float inv = 1.0f / Rs[wave][half * 8 + r];
      size_t obase = ((size_t)(b * SEQ + orow0 + r)) * DMODEL + h * DKV + l16;
#pragma unroll
      for (int nt = 0; nt < 8; ++nt)
        O[obase + nt * 16] = (__bf16)(acc[nt][r] * inv);
    }
  }
}

// ---------------------------------------------------------------------------
// host side
// ---------------------------------------------------------------------------
static inline void launch_cvt(const float* s, __bf16* d, size_t n,
                              hipStream_t stream) {
  cvt_f32_bf16<<<4096, 256, 0, stream>>>(s, d, n);
}

extern "C" void kernel_launch(void* const* d_in, const int* in_sizes, int n_in,
                              void* d_out, int out_size, void* d_ws,
                              size_t ws_size, hipStream_t stream) {
  const float* query = (const float*)d_in[0];
  const float* key   = (const float*)d_in[1];
  // d_in[2] (value) is ignored by the reference module.
  const float* Wq = (const float*)d_in[3];
  const float* bq = (const float*)d_in[4];
  const float* Wc = (const float*)d_in[5];
  const float* bc = (const float*)d_in[6];
  const float* Wk = (const float*)d_in[7];
  const float* bk = (const float*)d_in[8];
  const float* Wv = (const float*)d_in[9];
  const float* bv = (const float*)d_in[10];
  const float* Wo = (const float*)d_in[11];
  const float* bo = (const float*)d_in[12];

  char* ws = (char*)d_ws;
  size_t off = 0;
  auto alloc_bf = [&](size_t elems) -> __bf16* {
    __bf16* p = (__bf16*)(ws + off);
    off = (off + elems * sizeof(__bf16) + 255) & ~(size_t)255;
    return p;
  };

  __bf16* qbf  = alloc_bf((size_t)MROWS * DMODEL);
  __bf16* kbf  = alloc_bf((size_t)MROWS * DMODEL);
  __bf16* WqB  = alloc_bf((size_t)DMODEL * DMODEL);
  __bf16* WcB  = alloc_bf((size_t)DMODEL * LATENT);
  __bf16* WkB  = alloc_bf((size_t)LATENT * DMODEL);
  __bf16* WvB  = alloc_bf((size_t)LATENT * DMODEL);
  __bf16* WoB  = alloc_bf((size_t)DMODEL * DMODEL);
  __bf16* Qp   = alloc_bf((size_t)MROWS * DMODEL);
  __bf16* lat  = alloc_bf((size_t)MROWS * LATENT);
  __bf16* Kp   = alloc_bf((size_t)MROWS * DMODEL);
  __bf16* Vp   = alloc_bf((size_t)MROWS * DMODEL);
  __bf16* AO   = alloc_bf((size_t)MROWS * DMODEL);
  (void)ws_size; (void)n_in; (void)in_sizes; (void)out_size;

  launch_cvt(query, qbf, (size_t)MROWS * DMODEL, stream);
  launch_cvt(key,   kbf, (size_t)MROWS * DMODEL, stream);
  launch_cvt(Wq, WqB, (size_t)DMODEL * DMODEL, stream);
  launch_cvt(Wc, WcB, (size_t)DMODEL * LATENT, stream);
  launch_cvt(Wk, WkB, (size_t)LATENT * DMODEL, stream);
  launch_cvt(Wv, WvB, (size_t)LATENT * DMODEL, stream);
  launch_cvt(Wo, WoB, (size_t)DMODEL * DMODEL, stream);

  dim3 blk(256);
  // Q = q @ Wq + bq      [8192 x 2048 x 2048]
  gemm_bf16_wmma<<<dim3(DMODEL / 128, MROWS / 128), blk, 0, stream>>>(
      qbf, WqB, bq, nullptr, Qp, MROWS, DMODEL, DMODEL);
  // lat = k @ Wc + bc    [8192 x 512 x 2048]
  gemm_bf16_wmma<<<dim3(LATENT / 128, MROWS / 128), blk, 0, stream>>>(
      kbf, WcB, bc, nullptr, lat, MROWS, LATENT, DMODEL);
  // K = lat @ Wk + bk    [8192 x 2048 x 512]
  gemm_bf16_wmma<<<dim3(DMODEL / 128, MROWS / 128), blk, 0, stream>>>(
      lat, WkB, bk, nullptr, Kp, MROWS, DMODEL, LATENT);
  // V = lat @ Wv + bv    [8192 x 2048 x 512]
  gemm_bf16_wmma<<<dim3(DMODEL / 128, MROWS / 128), blk, 0, stream>>>(
      lat, WvB, bv, nullptr, Vp, MROWS, DMODEL, LATENT);

  // attention -> AO
  mla_attention<<<dim3(SEQ / 128, NHEAD, BATCH), blk, 0, stream>>>(Qp, Kp, Vp, AO);

  // out = AO @ Wo + bo   [8192 x 2048 x 2048], f32 to d_out
  gemm_bf16_wmma<<<dim3(DMODEL / 128, MROWS / 128), blk, 0, stream>>>(
      AO, WoB, bo, (float*)d_out, nullptr, MROWS, DMODEL, DMODEL);
}
